// RTMVSS6_64716567216794
// MI455X (gfx1250) — compile-verified, matrix operand order checked by reading, not run on hardware
//
#include <hip/hip_runtime.h>

typedef __attribute__((ext_vector_type(16))) _Float16 v16h;
typedef __attribute__((ext_vector_type(8)))  float    v8f;
typedef __attribute__((ext_vector_type(4)))  unsigned u32x4;
typedef __attribute__((ext_vector_type(8)))  unsigned u32x8;

#define DIMC   256
#define INNERC 64
#define NHC    8
#define HDC    8
#define BB     2
#define HWC    2304          // 48*48
#define NPOS   (BB * HWC)    // 4608

// ---------------------------------------------------------------------------
// WMMA helpers (CDNA5 / gfx1250, wave32)
// ---------------------------------------------------------------------------
__device__ __forceinline__ v8f wmma16(v16h a, v16h b, v8f c) {
  // D = A(16x32 f16) * B(32x16 f16) + C(16x16 f32)
  return __builtin_amdgcn_wmma_f32_16x16x32_f16(false, a, false, b, (short)0, c,
                                                false, false);
}

// Load a 16x32 f16 fragment in the CDNA5 A-layout.
// Logical element (r, k) lives at p + r*rstride + k*cstride.
// A-layout: lane 0-15 -> r = lane, k-base 0; lane 16-31 -> r = lane-16, k-base 8.
// frag[i]   = k = kbase + i        (i = 0..7)
// frag[8+i] = k = kbase + 16 + i
// B fragments (K x N) are loaded with the same code from the transposed view.
__device__ __forceinline__ v16h load_frag(const _Float16* __restrict__ p,
                                          long rstride, long cstride,
                                          int rvalid, int kvalid, int lane) {
  int r  = lane & 15;
  int kb = (lane >> 4) << 3;
  const _Float16* row = p + (long)r * rstride;
  bool rok = (r < rvalid);
  v16h f;
#pragma unroll
  for (int i = 0; i < 8; ++i) {
    int k0 = kb + i;
    int k1 = kb + 16 + i;
    f[i]     = (rok && k0 < kvalid) ? row[(long)k0 * cstride] : (_Float16)0.0f;
    f[i + 8] = (rok && k1 < kvalid) ? row[(long)k1 * cstride] : (_Float16)0.0f;
  }
  return f;
}

// Butterfly reductions. C-fragment rows are owned by a 16-lane half
// (lanes 0-15 hold rows 0-7, lanes 16-31 hold rows 8-15), so row-wise
// softmax reductions are xor-16 butterflies; LN uses the full 32 lanes.
__device__ __forceinline__ float redmax16(float v) {
#pragma unroll
  for (int off = 1; off < 16; off <<= 1) v = fmaxf(v, __shfl_xor(v, off, 32));
  return v;
}
__device__ __forceinline__ float redsum16(float v) {
#pragma unroll
  for (int off = 1; off < 16; off <<= 1) v += __shfl_xor(v, off, 32);
  return v;
}
__device__ __forceinline__ float redsum32(float v) {
#pragma unroll
  for (int off = 1; off < 32; off <<= 1) v += __shfl_xor(v, off, 32);
  return v;
}

// ---------------------------------------------------------------------------
// fp32 -> fp16 weight conversion
// ---------------------------------------------------------------------------
__global__ void rt_cvt(const float* __restrict__ s, _Float16* __restrict__ d,
                       int n) {
  int i = blockIdx.x * blockDim.x + threadIdx.x;
  if (i < n) d[i] = (_Float16)s[i];
}

// ---------------------------------------------------------------------------
// LayerNorm over C for NCHW input; writes position-major f16 normalized
// activations (WMMA A operand) and a raw f16 copy into the gate concat
// buffer cols [0,256).
// ---------------------------------------------------------------------------
__global__ void rt_ln_nchw(const float* __restrict__ x,   // [B, 256, HW]
                           const float* __restrict__ w,   // [256]
                           const float* __restrict__ b,   // [256]
                           _Float16* __restrict__ xnT,    // [NPOS, 256]
                           _Float16* __restrict__ rawT) { // [NPOS, 512], cols 0..255
  int pos  = blockIdx.x;             // 0..NPOS-1
  int bb   = pos / HWC;
  int hw   = pos % HWC;
  int lane = threadIdx.x;
  const float* xp = x + (size_t)bb * DIMC * HWC + hw;
  float v[8];
  float s = 0.f;
#pragma unroll
  for (int i = 0; i < 8; ++i) {
    v[i] = xp[(size_t)(i * 32 + lane) * HWC];
    s += v[i];
  }
  s = redsum32(s);
  float mean = s * (1.0f / DIMC);
  float var = 0.f;
#pragma unroll
  for (int i = 0; i < 8; ++i) {
    float d = v[i] - mean;
    var += d * d;
  }
  var = redsum32(var) * (1.0f / DIMC);
  float rstd = rsqrtf(var + 1e-6f);
#pragma unroll
  for (int i = 0; i < 8; ++i) {
    int c = i * 32 + lane;
    float xn = (v[i] - mean) * rstd * w[c] + b[c];
    xnT[(size_t)pos * DIMC + c]  = (_Float16)xn;
    rawT[(size_t)pos * 2 * DIMC + c] = (_Float16)v[i];
  }
}

// ---------------------------------------------------------------------------
// LayerNorm over contiguous rows (projection output), writing fp32 cross
// (for the residual) and f16 into gate concat cols [256,512).
// ---------------------------------------------------------------------------
__global__ void rt_ln_rows(const float* __restrict__ x,     // [NPOS, 256]
                           const float* __restrict__ w,
                           const float* __restrict__ b,
                           float* __restrict__ crossF,      // [NPOS, 256]
                           _Float16* __restrict__ gateA) {  // [NPOS, 512]
  int row  = blockIdx.x;
  int lane = threadIdx.x;
  const float* xp = x + (size_t)row * DIMC;
  float v[8];
  float s = 0.f;
#pragma unroll
  for (int i = 0; i < 8; ++i) {
    v[i] = xp[i * 32 + lane];
    s += v[i];
  }
  s = redsum32(s);
  float mean = s * (1.0f / DIMC);
  float var = 0.f;
#pragma unroll
  for (int i = 0; i < 8; ++i) {
    float d = v[i] - mean;
    var += d * d;
  }
  var = redsum32(var) * (1.0f / DIMC);
  float rstd = rsqrtf(var + 1e-6f);
#pragma unroll
  for (int i = 0; i < 8; ++i) {
    int c = i * 32 + lane;
    float xn = (v[i] - mean) * rstd * w[c] + b[c];
    crossF[(size_t)row * DIMC + c] = xn;
    gateA[(size_t)row * 2 * DIMC + DIMC + c] = (_Float16)xn;
  }
}

// ---------------------------------------------------------------------------
// WMMA GEMM with Tensor-Data-Mover weight staging.
//   C[M,Nout] = A[M,K](f16) * W[Nout,K](f16)^T + bias
// Block = 8 waves. Each block owns one 16-col N-tile and 8 consecutive 16-row
// M-tiles. Wave 0 issues one TDM transfer of the 16xK weight tile into LDS
// (D# groups built in SGPRs; 2D tile, data_size=2B), waits on TENSORcnt,
// then all waves stream A from global and B from LDS (8x weight reuse).
// ---------------------------------------------------------------------------
__global__ void __launch_bounds__(256)
rt_gemm_tdm(const _Float16* __restrict__ A, int lda,
            const _Float16* __restrict__ W, int ldw,   // W row-major [Nout,K]
            const float* __restrict__ bias,
            _Float16* __restrict__ outH,
            float* __restrict__ outF, int ldc,
            int Nout, int K) {
  __shared__ _Float16 wtile[16 * 512];   // 16 rows x K<=512 cols, 16 KB
  int wv   = (int)threadIdx.x >> 5;
  int lane = threadIdx.x & 31;
  int nt   = blockIdx.x;                 // N-tile
  int mt   = blockIdx.y * 8 + wv;        // M-tile per wave

  if (wv == 0) {
    // ---- Tensor DMA descriptor (D#) --------------------------------------
    unsigned long long ga =
        (unsigned long long)(const void*)(W + (size_t)nt * 16 * ldw);
    unsigned lds_off = (unsigned)(unsigned long long)(const void*)&wtile[0];
    unsigned k = (unsigned)K, n = (unsigned)Nout, st = (unsigned)ldw;
    u32x4 g0;
    g0[0] = 1u;                                         // count=1, user mode
    g0[1] = lds_off;                                    // lds_addr
    g0[2] = (unsigned)ga;                               // global_addr[31:0]
    g0[3] = (unsigned)((ga >> 32) & 0x1FFFFFFu)         // global_addr[56:32]
            | (2u << 30);                               // type = 2 (image)
    u32x8 g1;
    g1[0] = 1u << 16;                                   // data_size = 2 bytes
    g1[1] = (k & 0xFFFFu) << 16;                        // tensor_dim0[15:0] @48
    g1[2] = (k >> 16) | ((n & 0xFFFFu) << 16);          // dim0 hi | tensor_dim1 lo
    g1[3] = (n >> 16) | ((k & 0xFFFFu) << 16);          // dim1 hi | tile_dim0 = K
    g1[4] = 16u;                                        // tile_dim1=16, tile_dim2=0
    g1[5] = st;                                         // tensor_dim0_stride lo
    g1[6] = (st & 0xFFFFu) << 16;                       // stride0 hi(0)|stride1 lo
    g1[7] = 0u;                                         // stride1 hi
    asm volatile("tensor_load_to_lds %0, %1" : : "s"(g0), "s"(g1) : "memory");
    __builtin_amdgcn_s_wait_tensorcnt(0);
  }
  __syncthreads();

  const _Float16* Ab = A + (size_t)(mt * 16) * lda;
  v8f acc = {};
  for (int kk = 0; kk < K; kk += 32) {
    __builtin_prefetch(Ab + kk + 64, 0, 3);             // global_prefetch_b8
    v16h a = load_frag(Ab + kk, lda, 1, 16, 32, lane);  // global
    v16h b = load_frag(wtile + kk, K, 1, 16, 32, lane); // LDS (ds_load_b128)
    acc = wmma16(a, b, acc);
  }

  int col   = nt * 16 + (lane & 15);
  int rbase = mt * 16 + ((lane >> 4) << 3);
  float bv = bias ? bias[col] : 0.0f;
#pragma unroll
  for (int r = 0; r < 8; ++r) {
    float v = acc[r] + bv;
    size_t idx = (size_t)(rbase + r) * ldc + col;
    if (outF) outF[idx] = v;
    if (outH) outH[idx] = (_Float16)v;
  }
}

// ---------------------------------------------------------------------------
// Flash attention, one wave per (batch, head, 16-query tile).
// Q: [NPOS, 64] f16 ; KV: [NPOS, 128] f16 (k cols 0..63, v cols 64..127)
// HD = 8 -> the K dimension of Q.K^T is masked to 8 inside the fragment.
// P is staged through LDS to convert C-layout -> A-layout for P.V.
// ---------------------------------------------------------------------------
__global__ void rt_attn(const _Float16* __restrict__ Q,
                        const _Float16* __restrict__ KV,
                        _Float16* __restrict__ O) {
  __shared__ _Float16 pl[16][32];
  int qt   = blockIdx.x;        // 0..HWC/16-1
  int h    = blockIdx.y;        // head
  int b    = blockIdx.z;        // batch
  int lane = threadIdx.x;
  int half = lane >> 4;
  int c    = lane & 15;
  int hb   = h * HDC;
  int nbase = b * HWC;
  const _Float16* Qb  = Q  + (size_t)nbase * INNERC;
  const _Float16* Kb  = KV + (size_t)nbase * 2 * INNERC;                 // k
  const _Float16* Vbh = KV + (size_t)nbase * 2 * INNERC + INNERC + hb;   // v slice
  const float scale = 0.35355339059327373f;  // HD^-0.5

  v16h qa = load_frag(Qb + (size_t)(qt * 16) * INNERC + hb,
                      INNERC, 1, 16, HDC, lane);

  float rmax[8], rsum[8];
  v8f oacc = {};
#pragma unroll
  for (int r = 0; r < 8; ++r) { rmax[r] = -1e30f; rsum[r] = 0.f; }

  for (int m0 = 0; m0 < HWC; m0 += 32) {
    // S = Q K^T for 32 keys (two 16x16 tiles); B frag = A-style load of K rows.
    v16h kf0 = load_frag(Kb + (size_t)m0 * 2 * INNERC + hb,
                         2 * INNERC, 1, 16, HDC, lane);
    v16h kf1 = load_frag(Kb + (size_t)(m0 + 16) * 2 * INNERC + hb,
                         2 * INNERC, 1, 16, HDC, lane);
    v8f s0 = {}, s1 = {};
    s0 = wmma16(qa, kf0, s0);
    s1 = wmma16(qa, kf1, s1);
#pragma unroll
    for (int r = 0; r < 8; ++r) { s0[r] *= scale; s1[r] *= scale; }

    // Online softmax across the 32 new keys (row-wise over 16-lane halves).
#pragma unroll
    for (int r = 0; r < 8; ++r) {
      float mloc = redmax16(fmaxf(s0[r], s1[r]));
      float nm   = fmaxf(rmax[r], mloc);
      float corr = __expf(rmax[r] - nm);
      rmax[r] = nm;
      float p0 = __expf(s0[r] - nm);
      float p1 = __expf(s1[r] - nm);
      rsum[r] = rsum[r] * corr + redsum16(p0 + p1);
      oacc[r] *= corr;
      int row = r + 8 * half;            // C-layout row
      pl[row][c]      = (_Float16)p0;
      pl[row][16 + c] = (_Float16)p1;
    }
    __syncthreads();
    // O += P V  : A = P (16x32, from LDS), B = V (32 keys x 8 dims, masked)
    v16h pf = load_frag(&pl[0][0], 32, 1, 16, 32, lane);
    v16h vf = load_frag(Vbh + (size_t)m0 * 2 * INNERC,
                        1, 2 * INNERC, HDC, 32, lane);
    oacc = wmma16(pf, vf, oacc);
    __syncthreads();
  }

#pragma unroll
  for (int r = 0; r < 8; ++r) {
    float inv = 1.0f / rsum[r];
    if (c < HDC) {
      size_t n = (size_t)(nbase + qt * 16 + r + 8 * half);
      O[n * INNERC + hb + c] = (_Float16)(oacc[r] * inv);
    }
  }
}

// ---------------------------------------------------------------------------
// Residual: out = f + sigmoid(gate) * cross  (NCHW fp32)
// ---------------------------------------------------------------------------
__global__ void rt_final(const float* __restrict__ f,      // [B, 256, HW]
                         const float* __restrict__ cross,  // [NPOS, 256]
                         const float* __restrict__ gate,   // [NPOS, 256]
                         float* __restrict__ out) {
  int i = blockIdx.x * blockDim.x + threadIdx.x;
  int total = BB * DIMC * HWC;
  if (i >= total) return;
  int hw = i % HWC;
  int c  = (i / HWC) % DIMC;
  int b  = i / (HWC * DIMC);
  size_t n = (size_t)b * HWC + hw;
  float g = gate[n * DIMC + c];
  g = 1.0f / (1.0f + __expf(-g));
  out[i] = f[i] + g * cross[n * DIMC + c];
}

// ---------------------------------------------------------------------------
// Host launch
// ---------------------------------------------------------------------------
extern "C" void kernel_launch(void* const* d_in, const int* in_sizes, int n_in,
                              void* d_out, int out_size, void* d_ws,
                              size_t ws_size, hipStream_t stream) {
  (void)in_sizes; (void)n_in; (void)out_size; (void)ws_size;
  const float* f_rgb  = (const float*)d_in[0];
  const float* f_ir   = (const float*)d_in[1];
  const float* nr_w   = (const float*)d_in[2];
  const float* nr_b   = (const float*)d_in[3];
  const float* ni_w   = (const float*)d_in[4];
  const float* ni_b   = (const float*)d_in[5];
  const float* qr_w   = (const float*)d_in[6];
  const float* qr_b   = (const float*)d_in[7];
  const float* kvi_w  = (const float*)d_in[8];
  const float* kvi_b  = (const float*)d_in[9];
  const float* qi_w   = (const float*)d_in[10];
  const float* qi_b   = (const float*)d_in[11];
  const float* kvr_w  = (const float*)d_in[12];
  const float* kvr_b  = (const float*)d_in[13];
  const float* pr_w   = (const float*)d_in[14];
  const float* pr_b   = (const float*)d_in[15];
  const float* pr_lnw = (const float*)d_in[16];
  const float* pr_lnb = (const float*)d_in[17];
  const float* pi_w   = (const float*)d_in[18];
  const float* pi_b   = (const float*)d_in[19];
  const float* pi_lnw = (const float*)d_in[20];
  const float* pi_lnb = (const float*)d_in[21];
  const float* gr_w   = (const float*)d_in[22];
  const float* gr_b   = (const float*)d_in[23];
  const float* gi_w   = (const float*)d_in[24];
  const float* gi_b   = (const float*)d_in[25];

  char* p = (char*)d_ws;
  auto alloc = [&](size_t elems, size_t esz) -> char* {
    char* r = p;
    p += ((elems * esz) + 255) & ~(size_t)255;
    return r;
  };

  _Float16* wq_r  = (_Float16*)alloc((size_t)INNERC * DIMC, 2);
  _Float16* wkv_i = (_Float16*)alloc((size_t)2 * INNERC * DIMC, 2);
  _Float16* wq_i  = (_Float16*)alloc((size_t)INNERC * DIMC, 2);
  _Float16* wkv_r = (_Float16*)alloc((size_t)2 * INNERC * DIMC, 2);
  _Float16* wp_r  = (_Float16*)alloc((size_t)DIMC * INNERC, 2);
  _Float16* wp_i  = (_Float16*)alloc((size_t)DIMC * INNERC, 2);
  _Float16* wg_r  = (_Float16*)alloc((size_t)DIMC * 2 * DIMC, 2);
  _Float16* wg_i  = (_Float16*)alloc((size_t)DIMC * 2 * DIMC, 2);
  _Float16* n_rgb = (_Float16*)alloc((size_t)NPOS * DIMC, 2);
  _Float16* n_ir  = (_Float16*)alloc((size_t)NPOS * DIMC, 2);
  _Float16* gA_r  = (_Float16*)alloc((size_t)NPOS * 2 * DIMC, 2);
  _Float16* gA_i  = (_Float16*)alloc((size_t)NPOS * 2 * DIMC, 2);
  _Float16* q_r   = (_Float16*)alloc((size_t)NPOS * INNERC, 2);
  _Float16* kv_i  = (_Float16*)alloc((size_t)NPOS * 2 * INNERC, 2);
  _Float16* q_i   = (_Float16*)alloc((size_t)NPOS * INNERC, 2);
  _Float16* kv_r  = (_Float16*)alloc((size_t)NPOS * 2 * INNERC, 2);
  _Float16* ao_r  = (_Float16*)alloc((size_t)NPOS * INNERC, 2);
  _Float16* ao_i  = (_Float16*)alloc((size_t)NPOS * INNERC, 2);
  float* proj_r   = (float*)alloc((size_t)NPOS * DIMC, 4);
  float* proj_i   = (float*)alloc((size_t)NPOS * DIMC, 4);
  float* cross_r  = (float*)alloc((size_t)NPOS * DIMC, 4);
  float* cross_i  = (float*)alloc((size_t)NPOS * DIMC, 4);
  float* gate_r   = (float*)alloc((size_t)NPOS * DIMC, 4);
  float* gate_i   = (float*)alloc((size_t)NPOS * DIMC, 4);

  auto cvt = [&](const float* src, _Float16* dst, int n) {
    rt_cvt<<<dim3((n + 255) / 256), dim3(256), 0, stream>>>(src, dst, n);
  };
  auto gemm = [&](const _Float16* A, int lda, const _Float16* Wt, int ldw,
                  const float* bias, _Float16* oh, float* of, int ldc,
                  int M, int Nout, int K) {
    // grid: (N tiles, groups of 8 M tiles), block: 8 waves
    rt_gemm_tdm<<<dim3(Nout / 16, M / 128), dim3(256), 0, stream>>>(
        A, lda, Wt, ldw, bias, oh, of, ldc, Nout, K);
  };

  // Weights -> f16
  cvt(qr_w, wq_r, INNERC * DIMC);
  cvt(kvi_w, wkv_i, 2 * INNERC * DIMC);
  cvt(qi_w, wq_i, INNERC * DIMC);
  cvt(kvr_w, wkv_r, 2 * INNERC * DIMC);
  cvt(pr_w, wp_r, DIMC * INNERC);
  cvt(pi_w, wp_i, DIMC * INNERC);
  cvt(gr_w, wg_r, DIMC * 2 * DIMC);
  cvt(gi_w, wg_i, DIMC * 2 * DIMC);

  // LayerNorm + transpose + raw copy into gate concat buffers
  rt_ln_nchw<<<dim3(NPOS), dim3(32), 0, stream>>>(f_rgb, nr_w, nr_b, n_rgb, gA_r);
  rt_ln_nchw<<<dim3(NPOS), dim3(32), 0, stream>>>(f_ir, ni_w, ni_b, n_ir, gA_i);

  // q / kv projections (f16 out for attention)
  gemm(n_rgb, DIMC, wq_r, DIMC, qr_b, q_r, nullptr, INNERC, NPOS, INNERC, DIMC);
  gemm(n_ir, DIMC, wkv_i, DIMC, kvi_b, kv_i, nullptr, 2 * INNERC, NPOS,
       2 * INNERC, DIMC);
  gemm(n_ir, DIMC, wq_i, DIMC, qi_b, q_i, nullptr, INNERC, NPOS, INNERC, DIMC);
  gemm(n_rgb, DIMC, wkv_r, DIMC, kvr_b, kv_r, nullptr, 2 * INNERC, NPOS,
       2 * INNERC, DIMC);

  // Cross attention (rgb queries over ir kv, and vice versa)
  rt_attn<<<dim3(HWC / 16, NHC, BB), dim3(32), 0, stream>>>(q_r, kv_i, ao_r);
  rt_attn<<<dim3(HWC / 16, NHC, BB), dim3(32), 0, stream>>>(q_i, kv_r, ao_i);

  // Projection back to DIM (fp32 out for LN)
  gemm(ao_r, INNERC, wp_r, INNERC, pr_b, nullptr, proj_r, DIMC, NPOS, DIMC,
       INNERC);
  gemm(ao_i, INNERC, wp_i, INNERC, pi_b, nullptr, proj_i, DIMC, NPOS, DIMC,
       INNERC);

  // Post-projection LN -> cross (fp32) + gate concat cols [256,512)
  rt_ln_rows<<<dim3(NPOS), dim3(32), 0, stream>>>(proj_r, pr_lnw, pr_lnb,
                                                  cross_r, gA_r);
  rt_ln_rows<<<dim3(NPOS), dim3(32), 0, stream>>>(proj_i, pi_lnw, pi_lnb,
                                                  cross_i, gA_i);

  // Gate GEMM over concat(f, cross) : K = 512
  gemm(gA_r, 2 * DIMC, wg_r, 2 * DIMC, gr_b, nullptr, gate_r, DIMC, NPOS, DIMC,
       2 * DIMC);
  gemm(gA_i, 2 * DIMC, wg_i, 2 * DIMC, gi_b, nullptr, gate_i, DIMC, NPOS, DIMC,
       2 * DIMC);

  // Gated residual into the concatenated output (rgb first, then ir)
  float* out_rgb = (float*)d_out;
  float* out_ir  = out_rgb + (size_t)BB * DIMC * HWC;
  int total = BB * DIMC * HWC;
  rt_final<<<dim3((total + 255) / 256), dim3(256), 0, stream>>>(f_rgb, cross_r,
                                                                gate_r, out_rgb);
  rt_final<<<dim3((total + 255) / 256), dim3(256), 0, stream>>>(f_ir, cross_i,
                                                                gate_i, out_ir);
}
